// Aligner_6433861009562
// MI455X (gfx1250) — matrix operand order repaired and checked by reading, not compiled, not run on hardware
//
#include <hip/hip_runtime.h>
#include <hip/hip_bf16.h>

// ---------------------------------------------------------------------------
// Problem constants (from reference): B=32, L1=L2=512, D=1024
// ---------------------------------------------------------------------------
#define NB   32
#define NL1  512
#define NL2  512
#define ND   1024
#define MTOT (NB * (NL1 + NL2))   // 32768 projected rows total

typedef __attribute__((ext_vector_type(16))) __bf16 v16bf;
typedef __attribute__((ext_vector_type(8)))  float  v8f;

union FragBF {
    v16bf  v;
    uint4  q[2];
    __bf16 e[16];
};

static __device__ __forceinline__ __bf16 f2bf(float f) { return (__bf16)f; }

// ---------------------------------------------------------------------------
// Kernel 0: bulk fp32 -> bf16 conversion (8 elements / thread, b128 traffic)
// Used for ix -> XA[0:16384) rows and W -> Wbf.
// ---------------------------------------------------------------------------
__global__ __launch_bounds__(256) void cvt_kernel(
    const float* __restrict__ src, __bf16* __restrict__ dst, int n8)
{
    const int i = blockIdx.x * blockDim.x + threadIdx.x;   // group of 8 elems
    if (i >= n8) return;
    const float4* s = (const float4*)src + (size_t)i * 2;
    const float4 a = s[0];
    const float4 b = s[1];
    union { uint4 q; __bf16 e[8]; } u;
    u.e[0] = f2bf(a.x); u.e[1] = f2bf(a.y); u.e[2] = f2bf(a.z); u.e[3] = f2bf(a.w);
    u.e[4] = f2bf(b.x); u.e[5] = f2bf(b.y); u.e[6] = f2bf(b.z); u.e[7] = f2bf(b.w);
    ((uint4*)dst)[i] = u.q;
}

// ---------------------------------------------------------------------------
// Kernel 1: iother -> (a) row-major bf16 copy Obf (= XA second half) and
//                    (b) transposed bf16 Vt[b][d][m]  (LDS-tiled 32x32)
// ---------------------------------------------------------------------------
__global__ __launch_bounds__(256) void vt_kernel(
    const float* __restrict__ io, __bf16* __restrict__ Obf,
    __bf16* __restrict__ Vt)
{
    __shared__ float t[32][33];
    const int b  = blockIdx.z;
    const int mb = blockIdx.x * 32;
    const int db = blockIdx.y * 32;
    const int x  = threadIdx.x;   // 0..31
    const int y  = threadIdx.y;   // 0..7

#pragma unroll
    for (int i = 0; i < 4; ++i) {
        const int m = y + i * 8;
        const float v = io[((size_t)b * NL2 + mb + m) * ND + db + x];
        t[m][x] = v;
        Obf[((size_t)b * NL2 + mb + m) * ND + db + x] = f2bf(v);
    }
    __syncthreads();
#pragma unroll
    for (int i = 0; i < 4; ++i) {
        const int d = y + i * 8;
        Vt[((size_t)b * ND + db + d) * NL2 + mb + x] = f2bf(t[x][d]);
    }
}

// ---------------------------------------------------------------------------
// Kernel 2: shared linear projection  E = XA @ Wbf^T + b  (all-bf16 operands)
// XA = [ix ; iother] as bf16, 32768 x 1024. E likewise combined.
// WG = 256 threads = 8 waves. WG tile: 64 rows x 512 cols.
// Wave tile: 16 rows x 256 cols = 16 WMMA accumulators, K stepped by 32.
// Inner loop is pure global_load_b128 + v_wmma (no converts, no LDS).
// ---------------------------------------------------------------------------
__global__ __launch_bounds__(256) void proj_kernel(
    const __bf16* __restrict__ XA, const __bf16* __restrict__ Wbf,
    const float* __restrict__ bias, __bf16* __restrict__ E)
{
    const int tid    = threadIdx.x;
    const int lane   = tid & 31;
    const int w      = tid >> 5;
    const int lane15 = lane & 15;
    const bool hi    = lane >= 16;
    const int hi8    = hi ? 8 : 0;
    const int koff   = hi ? 8 : 0;    // A-fragment K sub-offset
    const int k16    = hi ? 16 : 0;   // unused; B fragment uses koff too

    const int rowBase = blockIdx.x * 64 + (w & 3) * 16;   // row in [0,32768)
    const int colBase = blockIdx.y * 512 + (w >> 2) * 256;

    const __bf16* arow = XA + (size_t)(rowBase + lane15) * ND;

    v8f acc[16] = {};

    for (int kb = 0; kb < ND; kb += 32) {
        FragBF a;
        const __bf16* pa = arow + kb + koff;
        a.q[0] = *(const uint4*)(pa);
        a.q[1] = *(const uint4*)(pa + 16);

#pragma unroll
        for (int nt = 0; nt < 16; ++nt) {
            // B[k][n] = W[colBase+nt*16+n][k] -> contiguous along d
            const __bf16* pb =
                Wbf + (size_t)(colBase + nt * 16 + lane15) * ND + kb + koff;
            FragBF bf_;
            bf_.q[0] = *(const uint4*)(pb);
            bf_.q[1] = *(const uint4*)(pb + 16);
            acc[nt] = __builtin_amdgcn_wmma_f32_16x16x32_bf16(
                false, a.v, false, bf_.v, (short)0, acc[nt], false, false);
        }
    }

    // bias + bf16 store (lanes 0..15 of each VGPR row form a contiguous 32B run)
#pragma unroll
    for (int nt = 0; nt < 16; ++nt) {
        const int col = colBase + nt * 16 + lane15;
        const float bv = bias[col];
#pragma unroll
        for (int r = 0; r < 8; ++r) {
            const int drow = rowBase + r + hi8;
            E[(size_t)drow * ND + col] = f2bf(acc[nt][r] + bv);
        }
    }
}

// ---------------------------------------------------------------------------
// Kernel 3: fused cross-attention per (batch, 64-row l1 tile).
//   Phase A: S = Ex_tile @ Eo^T  (WMMA, accums in regs, frags from L2)
//   write S (bf16) to LDS, softmax (max-sub, exp, rowsum kept separate),
//   Phase C: O = P @ V via Vt, scale by 1/rowsum, f32 store.
// 8 waves: wave = (rowGroup = w&3) x (colGroup = w>>2).
// ---------------------------------------------------------------------------
__global__ __launch_bounds__(256) void attn_kernel(
    const __bf16* __restrict__ Ex, const __bf16* __restrict__ Eo,
    const __bf16* __restrict__ Vt, float* __restrict__ out)
{
    __shared__ __bf16 S[64][NL2];        // 64 KB: scores, then P=exp(s-max)
    __shared__ float pmax[64][4];
    __shared__ float psum[64][4];

    const int tid    = threadIdx.x;
    const int lane   = tid & 31;
    const int w      = tid >> 5;
    const int lane15 = lane & 15;
    const bool hi    = lane >= 16;
    const int hi8    = hi ? 8 : 0;
    const int koff   = hi ? 8 : 0;
    const int k16    = hi ? 16 : 0;

    const int b   = blockIdx.y;
    const int l1b = blockIdx.x * 64;
    const int rb  = (w & 3) * 16;        // row block within the 64-row tile
    const int cb  = (w >> 2) * 256;      // col block over L2 for phase A

    // ---------------- Phase A: scores -------------------------------------
    {
        v8f sc[16] = {};
        const __bf16* arow = Ex + (size_t)(b * NL1 + l1b + rb + lane15) * ND;
        for (int kb = 0; kb < ND; kb += 32) {
            FragBF a;
            const __bf16* pa = arow + kb + koff;
            a.q[0] = *(const uint4*)(pa);
            a.q[1] = *(const uint4*)(pa + 16);
#pragma unroll
            for (int nt = 0; nt < 16; ++nt) {
                const __bf16* pb =
                    Eo + (size_t)(b * NL2 + cb + nt * 16 + lane15) * ND + kb + koff;
                FragBF bf_;
                bf_.q[0] = *(const uint4*)(pb);
                bf_.q[1] = *(const uint4*)(pb + 16);
                sc[nt] = __builtin_amdgcn_wmma_f32_16x16x32_bf16(
                    false, a.v, false, bf_.v, (short)0, sc[nt], false, false);
            }
        }
        // spill S (bf16) into LDS
#pragma unroll
        for (int nt = 0; nt < 16; ++nt) {
            const int col = cb + nt * 16 + lane15;
#pragma unroll
            for (int r = 0; r < 8; ++r)
                S[rb + r + hi8][col] = f2bf(sc[nt][r]);
        }
    }
    __syncthreads();

    // ---------------- Softmax: 4 threads per row --------------------------
    {
        const int row = tid >> 2;
        const int q   = tid & 3;
        float m = -3.0e38f;
        for (int j = 0; j < 128; ++j)
            m = fmaxf(m, (float)S[row][q * 128 + j]);
        pmax[row][q] = m;
        __syncthreads();
        const float rm = fmaxf(fmaxf(pmax[row][0], pmax[row][1]),
                               fmaxf(pmax[row][2], pmax[row][3]));
        float s = 0.f;
        for (int j = 0; j < 128; ++j) {
            const int c = q * 128 + j;
            const float e = __expf((float)S[row][c] - rm);
            S[row][c] = f2bf(e);
            s += e;
        }
        psum[row][q] = s;
    }
    __syncthreads();

    // ---------------- Phase C: O = P @ V  (via Vt), scale, store ----------
    float invs[8];
#pragma unroll
    for (int r = 0; r < 8; ++r) {
        const int lr = rb + r + hi8;
        const float s4 = psum[lr][0] + psum[lr][1] + psum[lr][2] + psum[lr][3];
        invs[r] = 1.0f / s4;
    }

    for (int pass = 0; pass < 2; ++pass) {
        const int cc = cb + pass * 512;   // output d-columns for this wave
        v8f oc[16] = {};
        const __bf16* parow = &S[rb + lane15][0];
        for (int kb = 0; kb < NL2; kb += 32) {
            FragBF a;
            a.q[0] = *(const uint4*)(parow + kb + koff);
            a.q[1] = *(const uint4*)(parow + kb + koff + 16);
#pragma unroll
            for (int nt = 0; nt < 16; ++nt) {
                // B[k=m][n=d] = Vt[b][d][m] -> contiguous along m
                const __bf16* pb =
                    Vt + (size_t)(b * ND + cc + nt * 16 + lane15) * NL2 + kb + koff;
                FragBF bf_;
                bf_.q[0] = *(const uint4*)(pb);
                bf_.q[1] = *(const uint4*)(pb + 16);
                oc[nt] = __builtin_amdgcn_wmma_f32_16x16x32_bf16(
                    false, a.v, false, bf_.v, (short)0, oc[nt], false, false);
            }
        }
#pragma unroll
        for (int nt = 0; nt < 16; ++nt) {
            const int col = cc + nt * 16 + lane15;
#pragma unroll
            for (int r = 0; r < 8; ++r) {
                const int orow = l1b + rb + r + hi8;
                out[((size_t)b * NL1 + orow) * ND + col] = oc[nt][r] * invs[r];
            }
        }
    }
}

// ---------------------------------------------------------------------------
// Host launcher
// ---------------------------------------------------------------------------
extern "C" void kernel_launch(void* const* d_in, const int* in_sizes, int n_in,
                              void* d_out, int out_size, void* d_ws, size_t ws_size,
                              hipStream_t stream) {
    const float* ix   = (const float*)d_in[0];   // [32,512,1024]
    const float* io   = (const float*)d_in[1];   // [32,512,1024]
    const float* W    = (const float*)d_in[2];   // [1024,1024]
    const float* bias = (const float*)d_in[3];   // [1024]
    float* out        = (float*)d_out;           // [32,512,1024] f32

    // Workspace layout (bf16):
    //   E  : [32768,1024]  combined Ex|Eo projections          (64 MB)
    //   XA : [32768,1024]  combined bf16(ix)|bf16(iother)      (64 MB)
    //   Vt : [32,1024,512] bf16 transpose of iother            (32 MB)
    //   Wbf: [1024,1024]                                        (2 MB)
    const size_t HALF_ROWS = (size_t)NB * NL1;            // 16384
    const size_t E_ELEMS   = (size_t)MTOT * ND;           // 33.5M
    char* ws = (char*)d_ws;
    __bf16* E   = (__bf16*)(ws);
    __bf16* XA  = (__bf16*)(ws + E_ELEMS * 2);
    __bf16* Vt  = (__bf16*)(ws + E_ELEMS * 4);
    __bf16* Wbf = (__bf16*)(ws + E_ELEMS * 4 + (size_t)NB * ND * NL2 * 2);
    (void)in_sizes; (void)n_in; (void)out_size; (void)ws_size;

    const __bf16* Ex = E;
    const __bf16* Eo = E + HALF_ROWS * ND;
    __bf16* Obf      = XA + HALF_ROWS * ND;

    // 0) one-time fp32 -> bf16 conversions
    {
        const int n8x = (int)(HALF_ROWS * ND / 8);        // ix elems / 8
        cvt_kernel<<<(n8x + 255) / 256, 256, 0, stream>>>(ix, XA, n8x);
        const int n8w = ND * ND / 8;                      // W elems / 8
        cvt_kernel<<<(n8w + 255) / 256, 256, 0, stream>>>(W, Wbf, n8w);
    }
    // 1) iother -> row-major bf16 (Obf) + transposed bf16 (Vt)
    vt_kernel<<<dim3(NL2 / 32, ND / 32, NB), dim3(32, 8), 0, stream>>>(io, Obf, Vt);
    // 2) shared linear projection of both sequences (all-bf16 WMMA GEMM)
    proj_kernel<<<dim3(MTOT / 64, ND / 512), 256, 0, stream>>>(XA, Wbf, bias, E);
    // 3) fused scores -> softmax -> attend
    attn_kernel<<<dim3(NL1 / 64, NB), 256, 0, stream>>>(Ex, Eo, Vt, out);
}
